// Net_43121471652562
// MI455X (gfx1250) — matrix operand order repaired and checked by reading, not compiled, not run on hardware
//
#include <hip/hip_runtime.h>
#include <math.h>

// ---------------------------------------------------------------------------
// MI455X (gfx1250) implementation: wave32, WMMA bf16 16x16x32, f32 accumulate.
// ---------------------------------------------------------------------------

typedef __attribute__((ext_vector_type(16))) __bf16 bf16x16;
typedef __attribute__((ext_vector_type(8)))  __bf16 bf16x8;
typedef __attribute__((ext_vector_type(8)))  float  v8f;

#define N_NODES 50000
#define N_EDGES 400000
#define F_IN    8
#define E_DIM   8
#define CH      64
#define KECC    288            // 256 (x ⊗ h) + 8 (bias rows) + 24 zero pad -> 9 k-steps of 32
#define TOT     (N_EDGES + N_NODES)

// orderable-uint mapping for float atomicMax (handles negatives)
__device__ __forceinline__ unsigned f2ord(float f) {
  unsigned u = __float_as_uint(f);
  return (u & 0x80000000u) ? ~u : (u | 0x80000000u);
}
__device__ __forceinline__ float ord2f(unsigned k) {
  unsigned u = (k & 0x80000000u) ? (k & 0x7FFFFFFFu) : ~k;
  return __uint_as_float(u);
}

// ---------------------------- utility fills --------------------------------
__global__ void fill_f32_kernel(float* p, int n, float v) {
  int i = blockIdx.x * blockDim.x + threadIdx.x;
  if (i < n) p[i] = v;
}
__global__ void fill_u32_kernel(unsigned* p, int n, unsigned v) {
  int i = blockIdx.x * blockDim.x + threadIdx.x;
  if (i < n) p[i] = v;
}

// ------------------- prep: build K-transposed bf16 B matrices ---------------
// BtEcc[o][k], k<256: W2[h, f*64+o] with k=f*32+h; k in [256,264): b2[f*64+o]; else 0
__global__ void prep_ecc_b_kernel(const float* __restrict__ w2,
                                  const float* __restrict__ b2,
                                  __bf16* __restrict__ BtEcc) {
  int tid = blockIdx.x * blockDim.x + threadIdx.x;
  if (tid >= CH * KECC) return;
  int o = tid / KECC, k = tid % KECC;
  float v = 0.0f;
  if (k < 256) {
    int f = k >> 5, hh = k & 31;
    v = w2[hh * (F_IN * CH) + f * CH + o];
  } else if (k < 264) {
    v = b2[(k - 256) * CH + o];
  }
  BtEcc[tid] = (__bf16)v;
}

// BtGat[o][k] = gat_kernel[k][o]
__global__ void prep_gat_b_kernel(const float* __restrict__ gk,
                                  __bf16* __restrict__ BtGat) {
  int tid = blockIdx.x * blockDim.x + threadIdx.x;
  if (tid >= CH * CH) return;
  int o = tid >> 6, k = tid & 63;
  BtGat[tid] = (__bf16)gk[k * CH + o];
}

// --------------------- ECC fused edge-MLP + GEMM + scatter ------------------
// Block = 64 threads = 2 waves; each wave owns 16 edges.
// A[e, f*32+h] = x_f * relu(E@w1+b1)_h ; A[e,256+f] = x_f ; pad to K=288.
// msg = A @ BtEcc^T accumulated with v_wmma_f32_16x16x32_bf16, scatter-added
// into agg[tgt] with global f32 atomics.
__global__ __launch_bounds__(64) void ecc_fused_kernel(
    const float* __restrict__ X, const float* __restrict__ Ef,
    const int* __restrict__ eidx, const float* __restrict__ w1,
    const float* __restrict__ b1, const __bf16* __restrict__ BtEcc,
    float* __restrict__ agg) {
  __shared__ __align__(64) __bf16 Bsh[CH * KECC];      // 36864 B
  __shared__ __align__(64) __bf16 Ash[2][16 * KECC];   // 2 x 9216 B
  __shared__ float w1sh[E_DIM * 32];
  __shared__ float b1sh[32];
  __shared__ int   tgtsh[2][16];

  const int tid = threadIdx.x;
  const int wv  = tid >> 5;
  const int ln  = tid & 31;

  // cooperative load of the shared B matrix (bf16, K-transposed) into LDS
  {
    const uint4* bs = (const uint4*)BtEcc;
    uint4* bd = (uint4*)Bsh;
    for (int i = tid; i < (CH * KECC) / 8; i += 64) bd[i] = bs[i];
    for (int i = tid; i < E_DIM * 32; i += 64) w1sh[i] = w1[i];
    if (tid < 32) b1sh[tid] = b1[tid];
  }
  __syncthreads();

  // phase 1: lanes 0..15 of each wave each build one A row (one edge)
  const int e0 = (blockIdx.x * 2 + wv) * 16;
  if (ln < 16) {
    const int e = e0 + ln;
    float ev[E_DIM];
#pragma unroll
    for (int d = 0; d < E_DIM; ++d) ev[d] = Ef[(size_t)e * E_DIM + d];
    const int sv = eidx[e];
    tgtsh[wv][ln] = eidx[N_EDGES + e];
    float xv[F_IN];
#pragma unroll
    for (int f = 0; f < F_IN; ++f) xv[f] = X[(size_t)sv * F_IN + f];
    __bf16* arow = &Ash[wv][ln * KECC];
#pragma unroll
    for (int hh = 0; hh < 32; ++hh) {
      float h = b1sh[hh];
#pragma unroll
      for (int d = 0; d < E_DIM; ++d) h += ev[d] * w1sh[d * 32 + hh];
      h = fmaxf(h, 0.0f);
#pragma unroll
      for (int f = 0; f < F_IN; ++f) arow[f * 32 + hh] = (__bf16)(xv[f] * h);
    }
#pragma unroll
    for (int f = 0; f < F_IN; ++f) arow[256 + f] = (__bf16)xv[f];
#pragma unroll
    for (int k = 264; k < KECC; ++k) arow[k] = (__bf16)0.0f;
  }
  __syncthreads();

  // phase 2: 9 k-steps x 4 n-tiles of WMMA
  const int am  = ln & 15;
  const int akb = (ln < 16) ? 0 : 16;   // A per-lane K base (contiguous K, matches B)
  const int bkb = (ln < 16) ? 0 : 16;   // B per-lane K base
  const v8f zero8 = {0.f, 0.f, 0.f, 0.f, 0.f, 0.f, 0.f, 0.f};
  v8f acc[4];
#pragma unroll
  for (int nt = 0; nt < 4; ++nt) acc[nt] = zero8;

  const __bf16* arow = &Ash[wv][am * KECC];
#pragma unroll
  for (int ks = 0; ks < KECC / 32; ++ks) {
    const int ko = ks * 32;
    bf16x8 alo = *(const bf16x8*)(arow + ko + akb);
    bf16x8 ahi = *(const bf16x8*)(arow + ko + akb + 8);
    bf16x16 a;
#pragma unroll
    for (int i = 0; i < 8; ++i) { a[i] = alo[i]; a[i + 8] = ahi[i]; }
#pragma unroll
    for (int nt = 0; nt < 4; ++nt) {
      bf16x16 b = *(const bf16x16*)&Bsh[(nt * 16 + am) * KECC + ko + bkb];
      acc[nt] = __builtin_amdgcn_wmma_f32_16x16x32_bf16(
          false, a, false, b, (short)0, acc[nt], false, false);
    }
  }

  // epilogue: scatter-add the 16x64 tile into agg[tgt] (C layout: VGPR i ->
  // row i (lanes 0-15) or i+8 (lanes 16-31), col = lane&15)
  const int mb = (ln < 16) ? 0 : 8;
#pragma unroll
  for (int nt = 0; nt < 4; ++nt) {
    const int chn = nt * 16 + am;
#pragma unroll
    for (int i = 0; i < 8; ++i) {
      const int t = tgtsh[wv][mb + i];
      atomicAdd(&agg[(size_t)t * CH + chn], acc[nt][i]);
    }
  }
}

// ---------------- x1 = relu(agg + X@root + bias), store bf16 ----------------
__global__ void x1_kernel(const float* __restrict__ agg,
                          const float* __restrict__ X,
                          const float* __restrict__ root,
                          const float* __restrict__ ebias,
                          __bf16* __restrict__ x1bf) {
  int tid = blockIdx.x * blockDim.x + threadIdx.x;
  if (tid >= N_NODES * CH) return;
  int c = tid & 63, n = tid >> 6;
  float a = agg[tid] + ebias[c];
#pragma unroll
  for (int f = 0; f < F_IN; ++f) a += X[(size_t)n * F_IN + f] * root[f * CH + c];
  x1bf[tid] = (__bf16)fmaxf(a, 0.0f);
}

// ---------------------- GAT GEMM: xg = x1 @ gat_kernel ----------------------
// One wave per block, 16 nodes per wave; A fragments loaded straight from
// global (rows contiguous), B (64x64 bf16, K-transposed) staged in LDS.
__global__ __launch_bounds__(32) void gat_gemm_kernel(
    const __bf16* __restrict__ x1bf, const __bf16* __restrict__ BtGat,
    float* __restrict__ xg) {
  __shared__ __align__(64) __bf16 Bsh[CH * CH];  // 8 KB
  const int ln = threadIdx.x & 31;
  {
    const uint4* bs = (const uint4*)BtGat;
    uint4* bd = (uint4*)Bsh;
    for (int i = ln; i < (CH * CH) / 8; i += 32) bd[i] = bs[i];
  }
  __syncthreads();

  const int n0  = blockIdx.x * 16;
  const int am  = ln & 15;
  const int akb = (ln < 16) ? 0 : 16;
  const int bkb = (ln < 16) ? 0 : 16;
  const v8f zero8 = {0.f, 0.f, 0.f, 0.f, 0.f, 0.f, 0.f, 0.f};
  v8f acc[4];
#pragma unroll
  for (int nt = 0; nt < 4; ++nt) acc[nt] = zero8;

  const __bf16* arow = x1bf + (size_t)(n0 + am) * CH;
#pragma unroll
  for (int ks = 0; ks < 2; ++ks) {
    const int ko = ks * 32;
    bf16x8 alo = *(const bf16x8*)(arow + ko + akb);
    bf16x8 ahi = *(const bf16x8*)(arow + ko + akb + 8);
    bf16x16 a;
#pragma unroll
    for (int i = 0; i < 8; ++i) { a[i] = alo[i]; a[i + 8] = ahi[i]; }
#pragma unroll
    for (int nt = 0; nt < 4; ++nt) {
      bf16x16 b = *(const bf16x16*)&Bsh[(nt * 16 + am) * CH + ko + bkb];
      acc[nt] = __builtin_amdgcn_wmma_f32_16x16x32_bf16(
          false, a, false, b, (short)0, acc[nt], false, false);
    }
  }

  const int mb = (ln < 16) ? 0 : 8;
#pragma unroll
  for (int nt = 0; nt < 4; ++nt) {
    const int chn = nt * 16 + am;
#pragma unroll
    for (int i = 0; i < 8; ++i)
      xg[(size_t)(n0 + mb + i) * CH + chn] = acc[nt][i];
  }
}

// -------------------- attention scalar scores per node ----------------------
__global__ void attn_scores_kernel(const float* __restrict__ xg,
                                   const float* __restrict__ as,
                                   const float* __restrict__ an,
                                   float* __restrict__ a_self,
                                   float* __restrict__ a_neigh) {
  int n = blockIdx.x * blockDim.x + threadIdx.x;
  if (n >= N_NODES) return;
  float s = 0.f, t = 0.f;
#pragma unroll 8
  for (int c = 0; c < CH; ++c) {
    float v = xg[(size_t)n * CH + c];
    s += v * as[c];
    t += v * an[c];
  }
  a_self[n] = s;
  a_neigh[n] = t;
}

// ------------------ logits + segment max (orderable atomicMax) --------------
__global__ void logits_max_kernel(const int* __restrict__ eidx,
                                  const float* __restrict__ a_self,
                                  const float* __restrict__ a_neigh,
                                  float* __restrict__ logits,
                                  unsigned* __restrict__ amax) {
  int i = blockIdx.x * blockDim.x + threadIdx.x;
  if (i >= TOT) return;
  int s = (i < N_EDGES) ? eidx[i] : (i - N_EDGES);
  int t = (i < N_EDGES) ? eidx[N_EDGES + i] : (i - N_EDGES);
  float v = a_self[t] + a_neigh[s];
  v = (v > 0.0f) ? v : 0.2f * v;  // leaky_relu(0.2)
  logits[i] = v;
  atomicMax(&amax[t], f2ord(v));
}

// ------------------------- exp + segment sum --------------------------------
__global__ void exp_sum_kernel(const int* __restrict__ eidx,
                               const float* __restrict__ logits,
                               const unsigned* __restrict__ amax,
                               float* __restrict__ exbuf,
                               float* __restrict__ z) {
  int i = blockIdx.x * blockDim.x + threadIdx.x;
  if (i >= TOT) return;
  int t = (i < N_EDGES) ? eidx[N_EDGES + i] : (i - N_EDGES);
  float ex = expf(logits[i] - ord2f(amax[t]));
  exbuf[i] = ex;
  atomicAdd(&z[t], ex);
}

// ------ attn = ex/z[t]; out2[t] += attn * xg[s]; attn -> output buffer ------
// 16 threads per edge, 4 channels each (float4).
__global__ void attn_scatter_kernel(const int* __restrict__ eidx,
                                    const float* __restrict__ exbuf,
                                    const float* __restrict__ z,
                                    const float* __restrict__ xg,
                                    float* __restrict__ out2,
                                    float* __restrict__ attn_out) {
  int tid = blockIdx.x * blockDim.x + threadIdx.x;
  if (tid >= TOT * 16) return;
  int i = tid >> 4, cg = tid & 15;
  int s = (i < N_EDGES) ? eidx[i] : (i - N_EDGES);
  int t = (i < N_EDGES) ? eidx[N_EDGES + i] : (i - N_EDGES);
  float a = exbuf[i] / z[t];
  if (cg == 0) attn_out[i] = a;
  float4 xv = ((const float4*)xg)[(size_t)s * 16 + cg];
  float* dst = &out2[(size_t)t * CH + cg * 4];
  atomicAdd(dst + 0, a * xv.x);
  atomicAdd(dst + 1, a * xv.y);
  atomicAdd(dst + 2, a * xv.z);
  atomicAdd(dst + 3, a * xv.w);
}

// --------------- x2 = relu(out2 + bias); mean pool (hierarchical) -----------
__global__ __launch_bounds__(256) void x2_pool_kernel(
    const float* __restrict__ out2, const float* __restrict__ gbias,
    float* __restrict__ poolsum) {
  __shared__ float sacc[CH];
  int tid = threadIdx.x;
  if (tid < CH) sacc[tid] = 0.0f;
  __syncthreads();
  int n = blockIdx.x * blockDim.x + tid;
  if (n < N_NODES) {
#pragma unroll 8
    for (int c = 0; c < CH; ++c) {
      float v = fmaxf(out2[(size_t)n * CH + c] + gbias[c], 0.0f);
      atomicAdd(&sacc[c], v);
    }
  }
  __syncthreads();
  if (tid < CH) atomicAdd(&poolsum[tid], sacc[tid]);
}

// ----------------------------- MLP head -------------------------------------
__global__ void head_kernel(const float* __restrict__ poolsum,
                            const float* __restrict__ fcw,
                            const float* __restrict__ fcb,
                            const float* __restrict__ outw,
                            const float* __restrict__ outb,
                            float* __restrict__ out) {
  __shared__ float fsh[32];
  int tid = threadIdx.x;
  if (tid < 32) {
    float acc = fcb[tid];
    const float inv = 1.0f / (float)N_NODES;
    for (int c = 0; c < CH; ++c) acc += poolsum[c] * inv * fcw[c * 32 + tid];
    fsh[tid] = fmaxf(acc, 0.0f);
  }
  __syncthreads();
  if (tid == 0) {
    float s = outb[0];
    for (int j = 0; j < 32; ++j) s += fsh[j] * outw[j];
    out[0] = 1.0f / (1.0f + expf(-s));
  }
}

// ---------------------------------------------------------------------------
extern "C" void kernel_launch(void* const* d_in, const int* in_sizes, int n_in,
                              void* d_out, int out_size, void* d_ws, size_t ws_size,
                              hipStream_t stream) {
  const float* X    = (const float*)d_in[0];
  const float* Ef   = (const float*)d_in[1];
  const int*   eidx = (const int*)  d_in[2];
  const float* w1   = (const float*)d_in[3];
  const float* b1   = (const float*)d_in[4];
  const float* w2   = (const float*)d_in[5];
  const float* b2   = (const float*)d_in[6];
  const float* root = (const float*)d_in[7];
  const float* ebias= (const float*)d_in[8];
  const float* gatk = (const float*)d_in[9];
  const float* atts = (const float*)d_in[10];
  const float* attnv= (const float*)d_in[11];
  const float* gbias= (const float*)d_in[12];
  const float* fcw  = (const float*)d_in[13];
  const float* fcb  = (const float*)d_in[14];
  const float* outw = (const float*)d_in[15];
  const float* outb = (const float*)d_in[16];
  float* out = (float*)d_out;            // [0]=sigmoid scalar, [1..TOT]=attn

  // workspace carve-up (256B aligned)
  char* wsp = (char*)d_ws;
  auto take = [&](size_t bytes) -> char* {
    char* p = wsp;
    wsp += (bytes + 255) & ~(size_t)255;
    return p;
  };
  __bf16*   BtEcc  = (__bf16*)  take((size_t)CH * KECC * 2);
  __bf16*   BtGat  = (__bf16*)  take((size_t)CH * CH * 2);
  float*    agg    = (float*)   take((size_t)N_NODES * CH * 4);
  __bf16*   x1bf   = (__bf16*)  take((size_t)(N_NODES + 16) * CH * 2);
  float*    xg     = (float*)   take((size_t)(N_NODES + 16) * CH * 4);
  float*    a_self = (float*)   take((size_t)N_NODES * 4);
  float*    a_neigh= (float*)   take((size_t)N_NODES * 4);
  unsigned* amax   = (unsigned*)take((size_t)N_NODES * 4);
  float*    z      = (float*)   take((size_t)N_NODES * 4);
  float*    logits = (float*)   take((size_t)TOT * 4);
  float*    exbuf  = (float*)   take((size_t)TOT * 4);
  float*    out2   = (float*)   take((size_t)N_NODES * CH * 4);
  float*    poolsum= (float*)   take(256);

  const int B = 256;
  // init accumulators (workspace is poisoned by the harness)
  fill_f32_kernel<<<(N_NODES * CH + B - 1) / B, B, 0, stream>>>(agg, N_NODES * CH, 0.0f);
  fill_f32_kernel<<<(N_NODES * CH + B - 1) / B, B, 0, stream>>>(out2, N_NODES * CH, 0.0f);
  fill_f32_kernel<<<(N_NODES + B - 1) / B, B, 0, stream>>>(z, N_NODES, 0.0f);
  fill_f32_kernel<<<1, 64, 0, stream>>>(poolsum, CH, 0.0f);
  fill_u32_kernel<<<(N_NODES + B - 1) / B, B, 0, stream>>>(amax, N_NODES, 0x007FFFFFu); // ord(-inf)

  // prepare bf16 K-transposed B matrices
  prep_ecc_b_kernel<<<(CH * KECC + B - 1) / B, B, 0, stream>>>(w2, b2, BtEcc);
  prep_gat_b_kernel<<<(CH * CH + B - 1) / B, B, 0, stream>>>(gatk, BtGat);

  // ECC: fused edge-MLP -> [400000 x 288] @ [288 x 64] WMMA -> atomic scatter
  ecc_fused_kernel<<<N_EDGES / 32, 64, 0, stream>>>(X, Ef, eidx, w1, b1, BtEcc, agg);

  // x1 = relu(agg + X@root + bias) -> bf16
  x1_kernel<<<(N_NODES * CH + B - 1) / B, B, 0, stream>>>(agg, X, root, ebias, x1bf);

  // GAT GEMM: xg = x1 @ gat_kernel (WMMA)
  gat_gemm_kernel<<<(N_NODES + 15) / 16, 32, 0, stream>>>(x1bf, BtGat, xg);

  // attention pipeline
  attn_scores_kernel<<<(N_NODES + B - 1) / B, B, 0, stream>>>(xg, atts, attnv, a_self, a_neigh);
  logits_max_kernel<<<(TOT + B - 1) / B, B, 0, stream>>>(eidx, a_self, a_neigh, logits, amax);
  exp_sum_kernel<<<(TOT + B - 1) / B, B, 0, stream>>>(eidx, logits, amax, exbuf, z);
  attn_scatter_kernel<<<((size_t)TOT * 16 + B - 1) / B, B, 0, stream>>>(
      eidx, exbuf, z, xg, out2, out + 1);

  // pool + head
  x2_pool_kernel<<<(N_NODES + B - 1) / B, B, 0, stream>>>(out2, gbias, poolsum);
  head_kernel<<<1, 64, 0, stream>>>(poolsum, fcw, fcb, outw, outb, out);
}
